// FMoELinearProj_3496103379132
// MI455X (gfx1250) — matrix operand order repaired and checked by reading, not compiled, not run on hardware
//
#include <hip/hip_runtime.h>

#define NE        16     // num experts
#define CAPACITY  1024   // static per-expert capacity
#define TM        128    // block tile M (tokens)
#define TN        128    // block tile N (outputs)
#define TK        32     // K step (bf16 WMMA K)
#define LDSS      40     // LDS row stride in bf16 elems (32 + 8 pad -> conflict-free b128)

typedef __attribute__((ext_vector_type(16))) __bf16 v16bf;
typedef __attribute__((ext_vector_type(8)))  __bf16 v8bf;
typedef __attribute__((ext_vector_type(4)))  __bf16 v4bf;
typedef __attribute__((ext_vector_type(8)))  float  v8f;

__device__ __forceinline__ __bf16 f2bf(float f) {
  // round-to-nearest-even fp32 -> bf16, pure integer
  union { float f; unsigned u; } c; c.f = f;
  unsigned u = c.u;
  unsigned r = (u + 0x7fffu + ((u >> 16) & 1u)) >> 16;
  union { unsigned short s; __bf16 b; } o; o.s = (unsigned short)r;
  return o.b;
}

// One GEMM kernel used for both stages.
//   STAGE1: A = inp fp32 [T,K], B = weight [E][N][K], +bias, out bf16 -> ws
//   STAGE2: A = y bf16 [T,K],   B = comps  [E][N][K],        out f32 -> d_out
template <bool STAGE1>
__global__ __launch_bounds__(256, 2)
void moe_wmma_gemm(const float*  __restrict__ Af,
                   const __bf16* __restrict__ Ab,
                   const float*  __restrict__ B,
                   const float*  __restrict__ bias,
                   const int*    __restrict__ counts,
                   __bf16*       __restrict__ Ob,
                   float*        __restrict__ Of,
                   int T, int K, int N)
{
  const int e = blockIdx.z;
  int off = 0, cnt = 0;
  #pragma unroll
  for (int i = 0; i < NE; ++i) {               // tiny scalar prefix sum (E=16)
    int c = counts[i];
    if (i < e) off += c;
    if (i == e) cnt = c;
  }
  const int mBase = blockIdx.y * TM;
  if (mBase >= cnt) return;                    // uniform early-exit past expert's count
  const int nBase     = blockIdx.x * TN;
  const int rowBase   = off + mBase;           // global token row of this tile
  const int rowsValid = (cnt - mBase < TM) ? (cnt - mBase) : TM;

  // double-buffered LDS (ping-pong): stage k+1 while WMMAs consume k
  __shared__ __bf16 lA[2][TM * LDSS];
  __shared__ __bf16 lB[2][TN * LDSS];

  const int tid  = threadIdx.x;
  const int lane = tid & 31;
  const int wid  = tid >> 5;
  const int wm   = (wid >> 2) * 64;            // 2 waves along M (64 rows each)
  const int wn   = (wid & 3) * 32;             // 4 waves along N (32 cols each)
  const int hl   = lane >> 4;                  // half-wave select
  const int l16  = lane & 15;

  const float* Be = B + (size_t)e * (size_t)N * (size_t)K;

  // ---- tile stager: global (fp32 or bf16) -> LDS bf16 ----
  auto stage = [&](int k0, __bf16* SA, __bf16* SB) {
    #pragma unroll
    for (int p = 0; p < 4; ++p) {
      const int idx = p * 256 + tid;
      const int r   = idx >> 3;                // tile row 0..127
      const int kc  = (idx & 7) << 2;          // k offset 0,4,...,28
      int grow = rowBase + r;                  // clamp; garbage rows masked at store
      if (grow > T - 1) grow = T - 1;
      __bf16* da = &SA[r * LDSS + kc];
      if (STAGE1) {
        const float4 va = *(const float4*)(Af + (size_t)grow * K + (k0 + kc));
        da[0] = f2bf(va.x); da[1] = f2bf(va.y); da[2] = f2bf(va.z); da[3] = f2bf(va.w);
      } else {
        *(v4bf*)da = *(const v4bf*)(Ab + (size_t)grow * K + (k0 + kc));
      }
      const float4 vb = *(const float4*)(Be + (size_t)(nBase + r) * K + (k0 + kc));
      __bf16* db = &SB[r * LDSS + kc];
      db[0] = f2bf(vb.x); db[1] = f2bf(vb.y); db[2] = f2bf(vb.z); db[3] = f2bf(vb.w);
    }
  };

  v8f acc[4][2];
  #pragma unroll
  for (int mi = 0; mi < 4; ++mi)
    #pragma unroll
    for (int ni = 0; ni < 2; ++ni)
      acc[mi][ni] = {};

  stage(0, lA[0], lB[0]);                      // prologue fill of buffer 0

  const int nIters = K / TK;
  for (int it = 0; it < nIters; ++it) {
    // Orders: (a) staging of buffer `cur` (prev iter) before its frag reads,
    //         (b) frag reads of buffer `cur^1` (prev iter) before its overwrite.
    __syncthreads();
    const int cur = it & 1;
    if (it + 1 < nIters)
      stage((it + 1) * TK, lA[cur ^ 1], lB[cur ^ 1]);   // overlap with WMMAs below

    // ---- fragment loads per ISA 7.12.2 layouts, then 8 WMMAs ----
    const __bf16* cA = lA[cur];
    const __bf16* cB = lB[cur];
    v16bf a[4], b[2];
    #pragma unroll
    for (int mi = 0; mi < 4; ++mi) {
      // A 16x32: lane m=l16; low half-wave K{0..7,16..23}, high K{8..15,24..31}
      const __bf16* pa = &cA[(wm + mi * 16 + l16) * LDSS + hl * 8];
      v8bf lo = *(const v8bf*)pa;
      v8bf hi = *(const v8bf*)(pa + 16);
      a[mi] = __builtin_shufflevector(lo, hi, 0,1,2,3,4,5,6,7,8,9,10,11,12,13,14,15);
    }
    #pragma unroll
    for (int ni = 0; ni < 2; ++ni) {
      // B 32x16: lane n=l16; low half-wave K{0..7,16..23}, high K{8..15,24..31}
      const __bf16* pb = &cB[(wn + ni * 16 + l16) * LDSS + hl * 8];
      v8bf lo = *(const v8bf*)pb;
      v8bf hi = *(const v8bf*)(pb + 16);
      b[ni] = __builtin_shufflevector(lo, hi, 0,1,2,3,4,5,6,7,8,9,10,11,12,13,14,15);
    }
    #pragma unroll
    for (int mi = 0; mi < 4; ++mi)
      #pragma unroll
      for (int ni = 0; ni < 2; ++ni)
        acc[mi][ni] = __builtin_amdgcn_wmma_f32_16x16x32_bf16(
            false, a[mi], false, b[ni], (short)0, acc[mi][ni], false, false);
  }

  // ---- epilogue: C layout VGPR r -> M = r + (lane>=16)*8, N = lane&15 ----
  if (rowsValid == TM) {
    // fast path: full tile (balanced counts) -> no per-row masking
    #pragma unroll
    for (int ni = 0; ni < 2; ++ni) {
      const int col = nBase + wn + ni * 16 + l16;
      const float bv = STAGE1 ? bias[e * N + col] : 0.0f;
      #pragma unroll
      for (int mi = 0; mi < 4; ++mi) {
        const size_t base = (size_t)(rowBase + wm + mi * 16 + hl * 8) * (size_t)N + col;
        #pragma unroll
        for (int r = 0; r < 8; ++r) {
          if (STAGE1) Ob[base + (size_t)r * N] = f2bf(acc[mi][ni][r] + bv);
          else        Of[base + (size_t)r * N] = acc[mi][ni][r];
        }
      }
    }
  } else {
    #pragma unroll
    for (int ni = 0; ni < 2; ++ni) {
      const int col = nBase + wn + ni * 16 + l16;
      const float bv = STAGE1 ? bias[e * N + col] : 0.0f;
      #pragma unroll
      for (int mi = 0; mi < 4; ++mi) {
        const int lmBase = wm + mi * 16 + hl * 8;
        #pragma unroll
        for (int r = 0; r < 8; ++r) {
          const int lm = lmBase + r;
          if (lm < rowsValid) {
            const size_t o = (size_t)(rowBase + lm) * (size_t)N + col;
            if (STAGE1) Ob[o] = f2bf(acc[mi][ni][r] + bv);
            else        Of[o] = acc[mi][ni][r];
          }
        }
      }
    }
  }
}

extern "C" void kernel_launch(void* const* d_in, const int* in_sizes, int n_in,
                              void* d_out, int out_size, void* d_ws, size_t ws_size,
                              hipStream_t stream) {
  const float* inp    = (const float*)d_in[0];
  const int*   counts = (const int*)d_in[1];
  const float* weight = (const float*)d_in[2];
  const float* bias   = (const float*)d_in[3];
  const float* comps  = (const float*)d_in[4];
  float*  out = (float*)d_out;
  __bf16* yws = (__bf16*)d_ws;                 // intermediate Y, bf16 [T, Dout] = 64 MB

  const int Dout = in_sizes[3] / NE;           // 4096
  const int Din  = in_sizes[2] / (NE * Dout);  // 1024
  const int S    = in_sizes[4] / (NE * Dout);  // 512
  const int T    = in_sizes[0] / Din;          // 8192

  dim3 blk(256);
  // Stage 1: Y = X @ W^T + b   (per expert), bf16 result into workspace
  dim3 g1(Dout / TN, CAPACITY / TM, NE);
  moe_wmma_gemm<true><<<g1, blk, 0, stream>>>(inp, nullptr, weight, bias, counts,
                                              yws, nullptr, T, Din, Dout);
  // Stage 2: out = Y @ C^T     (per expert), f32 result
  dim3 g2(S / TN, CAPACITY / TM, NE);
  moe_wmma_gemm<false><<<g2, blk, 0, stream>>>(nullptr, yws, comps, nullptr, counts,
                                               nullptr, out, T, Dout, S);
}
